// SQLDecoder_43138651521703
// MI455X (gfx1250) — matrix operand order repaired and checked by reading, not compiled, not run on hardware
//
#include <hip/hip_runtime.h>
#include <hip/hip_bf16.h>
#include <stdint.h>

// ---------------------------------------------------------------------------
// SQL sketch decoder for MI455X (gfx1250).
//   B=64 S=1024 T=64 H=1024 A=1024 DA=512 DT=256 P=256 NT=32 DIN=1792
// Strategy:
//   * refactor scores = (h @ W_att) . enc  -> no 256MB src_att tensor
//   * single-pass online-softmax attention streaming enc once/step (16GB total)
//     staged via GLOBAL_LOAD_ASYNC_TO_LDS_B128, double-buffered (ASYNCcnt)
//   * all dense math in bf16 WMMA (v_wmma_f32_16x16x32_bf16), fp32 accum
// ---------------------------------------------------------------------------

typedef __bf16 bf16_t;
typedef bf16_t bf16x8  __attribute__((ext_vector_type(8)));
typedef bf16_t bf16x16 __attribute__((ext_vector_type(16)));
typedef float  f32x8   __attribute__((ext_vector_type(8)));

#define C_B   64
#define C_S   1024
#define C_T   64
#define C_H   1024
#define C_A   1024
#define C_DA  512
#define C_DT  256
#define C_P   256
#define C_DIN 1792
#define C_KCAT (C_DIN + C_H)   // 2816
#define C_G4H (4 * C_H)        // 4096

__device__ __forceinline__ float sigm(float x) { return 1.0f / (1.0f + __expf(-x)); }

// ---------------------------------------------------------------------------
// Generic WMMA bf16 GEMM:  D[M,N] = act( A[M,K] @ Bm[N,K]^T + bias[N] )
// One 16x16 output tile per wave32; 4 waves per block.
// A-layout per ISA 7.12.2: lane(l) row = l&15, half = l>>4,
//   vector elem e -> k = half*8 + e (e<8) | half*8 + e + 8 (e>=8)
// B-layout (column-major per lane): lane col = l&15, elem e -> same k as A
// ---------------------------------------------------------------------------
__global__ void __launch_bounds__(128)
wmma_gemm_bf16(const bf16_t* __restrict__ A, int lda,
               const bf16_t* __restrict__ Bm, int ldb,
               const float* __restrict__ bias,
               float* __restrict__ D, bf16_t* __restrict__ Dbf, int ldd,
               int M, int N, int K, int act)
{
    const int wave = threadIdx.x >> 5;
    const int lane = threadIdx.x & 31;
    const int ntn  = N >> 4;
    const int tile = blockIdx.x * 4 + wave;
    const int tm   = (tile / ntn) << 4;
    const int tn   = (tile % ntn) << 4;
    if (tm >= M) return;                      // wave-uniform guard

    const int half = lane >> 4;
    const int mrow = lane & 15;               // A row within tile
    const int ncol = lane & 15;               // B column within tile

    const bf16_t* Ap = A  + (size_t)(tm + mrow) * lda + half * 8;
    const bf16_t* Bp = Bm + (size_t)(tn + ncol) * ldb + half * 8;

    f32x8 acc = {};
    for (int kk = 0; kk < K; kk += 32) {
        bf16x8 alo = *(const bf16x8*)(Ap + kk);        // k = half*8 + [0..7]
        bf16x8 ahi = *(const bf16x8*)(Ap + kk + 16);   // k = half*8 + 16 + [0..7]
        bf16x16 av = __builtin_shufflevector(alo, ahi,
                        0,1,2,3,4,5,6,7,8,9,10,11,12,13,14,15);
        bf16x8 blo = *(const bf16x8*)(Bp + kk);        // k = half*8 + [0..7]
        bf16x8 bhi = *(const bf16x8*)(Bp + kk + 16);   // k = half*8 + 16 + [0..7]
        bf16x16 bv = __builtin_shufflevector(blo, bhi,
                        0,1,2,3,4,5,6,7,8,9,10,11,12,13,14,15);
        acc = __builtin_amdgcn_wmma_f32_16x16x32_bf16(
                  false, av, false, bv, (short)0, acc, false, false);
    }

    // C/D layout: lane n = tn + (l&15); VGPR r -> m = tm + half*8 + r
    const int gn = tn + ncol;
    const float bb = bias ? bias[gn] : 0.0f;
#pragma unroll
    for (int r = 0; r < 8; ++r) {
        const int gm = tm + half * 8 + r;
        float v = acc[r] + bb;
        if (act == 1) v = tanhf(v);
        if (D)   D[(size_t)gm * ldd + gn]   = v;
        if (Dbf) Dbf[(size_t)gm * ldd + gn] = (bf16_t)v;
    }
}

// --------------------------- weight preparation ----------------------------
__global__ void convert_f32_bf16(const float* __restrict__ src,
                                 bf16_t* __restrict__ dst, int n)
{
    int i = blockIdx.x * 256 + threadIdx.x;
    if (i < n) dst[i] = (bf16_t)src[i];
}

__global__ void build_wcat(const float* __restrict__ Wih,
                           const float* __restrict__ Whh,
                           bf16_t* __restrict__ Wcat, int n)
{
    int i = blockIdx.x * 256 + threadIdx.x;
    if (i >= n) return;
    int r = i / C_KCAT, c = i % C_KCAT;
    float v = (c < C_DIN) ? Wih[(size_t)r * C_DIN + c]
                          : Whh[(size_t)r * C_H + (c - C_DIN)];
    Wcat[i] = (bf16_t)v;
}

__global__ void transpose_convert(const float* __restrict__ W,
                                  bf16_t* __restrict__ WT, int rows, int cols)
{
    int i = blockIdx.x * 256 + threadIdx.x;
    if (i >= rows * cols) return;
    int r = i / cols, c = i % cols;
    WT[(size_t)c * rows + r] = (bf16_t)W[i];
}

__global__ void bias_sum(const float* __restrict__ a, const float* __restrict__ b,
                         float* __restrict__ o, int n)
{
    int i = blockIdx.x * 256 + threadIdx.x;
    if (i < n) o[i] = a[i] + b[i];
}

__global__ void zero_init(float* __restrict__ c, bf16_t* __restrict__ att,
                          float* __restrict__ out)
{
    int i = blockIdx.x * 256 + threadIdx.x;
    if (i < C_B * C_H) { c[i] = 0.0f; att[i] = (bf16_t)0.0f; }
    if (i < C_B) out[i] = 0.0f;
}

__global__ void mean_kernel(const float* __restrict__ enc,
                            bf16_t* __restrict__ mean_bf)
{
    int b = blockIdx.y;
    int h = blockIdx.x * 256 + threadIdx.x;
    const float* p = enc + (size_t)b * C_S * C_H + h;
    float s = 0.0f;
    for (int i = 0; i < C_S; ++i) s += p[(size_t)i * C_H];
    mean_bf[b * C_H + h] = (bf16_t)(s * (1.0f / C_S));
}

// --------------------------- per-step kernels ------------------------------
// xh = [ a_embed(512) | att_tm1(1024) | type_embed(256) | h(1024) ] in bf16
__global__ void __launch_bounds__(256)
assemble_xh(const int* __restrict__ prev_pid, const int* __restrict__ prev_tid,
            const int* __restrict__ sk_lens,
            const float* __restrict__ prod_embed, const float* __restrict__ type_embed,
            const bf16_t* __restrict__ att_bf, const bf16_t* __restrict__ h_bf,
            bf16_t* __restrict__ xh, int t)
{
    int b   = blockIdx.y;
    int col = blockIdx.x * 256 + threadIdx.x;   // 2816 = 11 * 256
    bf16_t v;
    if (col >= C_DIN) {
        v = h_bf[b * C_H + (col - C_DIN)];
    } else if (t == 0) {
        v = (bf16_t)0.0f;                       // x zeroed at t==0
    } else {
        bool pv = (t < sk_lens[b]);             // prev_valid (t>0 here)
        if (col < C_DA) {
            v = pv ? (bf16_t)prod_embed[(size_t)prev_pid[b * C_T + t] * C_DA + col]
                   : (bf16_t)0.0f;
        } else if (col < C_DA + C_A) {
            v = att_bf[b * C_A + (col - C_DA)];
        } else {
            v = pv ? (bf16_t)type_embed[(size_t)prev_tid[b * C_T + t] * C_DT + (col - C_DA - C_A)]
                   : (bf16_t)0.0f;
        }
    }
    xh[(size_t)b * C_KCAT + col] = v;
}

__global__ void __launch_bounds__(256)
lstm_pointwise(const float* __restrict__ gates, float* __restrict__ c,
               bf16_t* __restrict__ h_bf, bf16_t* __restrict__ hc_bf)
{
    int b = blockIdx.y;
    int h = blockIdx.x * 256 + threadIdx.x;
    const float* g = gates + (size_t)b * C_G4H;
    float gi = g[h], gf = g[h + C_H], gg = g[h + 2 * C_H], go = g[h + 3 * C_H];
    float cn = sigm(gf) * c[b * C_H + h] + sigm(gi) * tanhf(gg);
    float hn = sigm(go) * tanhf(cn);
    c[b * C_H + h] = cn;
    bf16_t hb = (bf16_t)hn;
    h_bf[b * C_H + h] = hb;
    hc_bf[(size_t)b * (2 * C_H) + h] = hb;      // [h | ctx] for att-vec GEMM
}

// ---------------------------------------------------------------------------
// Fused attention: one block per batch row; online softmax over S.
// enc chunks are staged into LDS with GLOBAL_LOAD_ASYNC_TO_LDS_B128
// (ASYNCcnt path, no VGPR transit), double-buffered: chunk c+1 is in flight
// while chunk c is consumed. Per wave, each chunk = 8 async ops, so
// s_wait_asynccnt 0x8 releases the current chunk while the next streams.
// ---------------------------------------------------------------------------
#define ACS 8
__global__ void __launch_bounds__(256)
attention_kernel(const float* __restrict__ enc, const float* __restrict__ hW,
                 const int* __restrict__ src_lens, bf16_t* __restrict__ hc_bf)
{
    __shared__ float sEnc[2][ACS][C_H];  // 2 x 32 KB double buffer
    __shared__ float sHW[C_H];           // 4 KB
    __shared__ float sSc[ACS];
    __shared__ float sW[ACS];
    __shared__ float sR;
    __shared__ float sInvL;

    const int b    = blockIdx.x;
    const int tid  = threadIdx.x;
    const int slen = src_lens[b];
    const float* encB = enc + (size_t)b * C_S * C_H;

    for (int i = tid; i < C_H; i += 256) sHW[i] = hW[(size_t)b * C_H + i];

    float accv[4] = {0.f, 0.f, 0.f, 0.f};      // thread owns h = tid + k*256
    float m = -3.0e38f, l = 0.0f;              // only thread 0's copies live
    __syncthreads();

    // async stage one 32KB chunk (ACS rows) into sEnc[buf]
    auto issue_chunk = [&](int s0, int buf) {
        const float4* src4 = (const float4*)(encB + (size_t)s0 * C_H);
        const unsigned lbase = (unsigned)(uintptr_t)(&sEnc[buf][0][0]);
#pragma unroll
        for (int j = 0; j < ACS * (C_H / 4) / 256; ++j) {       // 8 per thread
            const int i = tid + j * 256;
            const unsigned lds_addr = lbase + (unsigned)i * 16u;
            const unsigned long long gaddr = (unsigned long long)(src4 + i);
            asm volatile("global_load_async_to_lds_b128 %0, %1, off"
                         :: "v"(lds_addr), "v"(gaddr) : "memory");
        }
    };

    issue_chunk(0, 0);
    const int nchunks = C_S / ACS;             // 128
    for (int c = 0; c < nchunks; ++c) {
        const int buf = c & 1;
        const int s0  = c * ACS;
        if (c + 1 < nchunks) {
            issue_chunk(s0 + ACS, buf ^ 1);    // prefetch next chunk
            asm volatile("s_wait_asynccnt 0x8" ::: "memory");
        } else {
            asm volatile("s_wait_asynccnt 0x0" ::: "memory");
        }
        __syncthreads();                       // all waves' chunk-c data in LDS

        // scores: wave w computes row s0+w (wave32 shuffle reduction)
        {
            const int w = tid >> 5, lane = tid & 31;
            float p = 0.0f;
            const float* row = sEnc[buf][w];
            for (int j = lane; j < C_H; j += 32) p += sHW[j] * row[j];
#pragma unroll
            for (int off = 16; off; off >>= 1) p += __shfl_xor(p, off, 32);
            if (lane == 0) {
                int s = s0 + w;
                sSc[w] = (s < slen) ? p : -1e9f;
            }
        }
        __syncthreads();

        if (tid == 0) {                        // online softmax bookkeeping
            float cmax = sSc[0];
#pragma unroll
            for (int i = 1; i < ACS; ++i) cmax = fmaxf(cmax, sSc[i]);
            float mn = fmaxf(m, cmax);
            float r  = __expf(m - mn);         // first iter: exp(-huge)=0
            float sum = 0.0f;
#pragma unroll
            for (int i = 0; i < ACS; ++i) { float e = __expf(sSc[i] - mn); sW[i] = e; sum += e; }
            l = l * r + sum;
            m = mn;
            sR = r;
        }
        __syncthreads();

        {
            const float r = sR;
#pragma unroll
            for (int k = 0; k < 4; ++k) {
                const int h = tid + k * 256;
                float a = accv[k] * r;
#pragma unroll
                for (int i = 0; i < ACS; ++i) a += sW[i] * sEnc[buf][i][h];
                accv[k] = a;
            }
        }
        __syncthreads();   // buf is refilled by issue_chunk in the next iteration
    }

    if (tid == 0) sInvL = 1.0f / l;
    __syncthreads();
    const float inv = sInvL;
#pragma unroll
    for (int k = 0; k < 4; ++k) {
        const int h = tid + k * 256;
        hc_bf[(size_t)b * (2 * C_H) + C_H + h] = (bf16_t)(accv[k] * inv);
    }
}

__global__ void __launch_bounds__(256)
softmax_logp_kernel(const float* __restrict__ logits, const int* __restrict__ tgt,
                    const int* __restrict__ sk, float* __restrict__ out, int t)
{
    __shared__ float red[8];
    __shared__ float sMax;
    const int b = blockIdx.x, tid = threadIdx.x;   // P = 256 threads
    const float x = logits[b * C_P + tid];

    float v = x;
#pragma unroll
    for (int off = 16; off; off >>= 1) v = fmaxf(v, __shfl_xor(v, off, 32));
    if ((tid & 31) == 0) red[tid >> 5] = v;
    __syncthreads();
    if (tid == 0) {
        float mm = red[0];
#pragma unroll
        for (int i = 1; i < 8; ++i) mm = fmaxf(mm, red[i]);
        sMax = mm;
    }
    __syncthreads();

    float e = __expf(x - sMax);
#pragma unroll
    for (int off = 16; off; off >>= 1) e += __shfl_xor(e, off, 32);
    if ((tid & 31) == 0) red[tid >> 5] = e;
    __syncthreads();
    if (tid == 0) {
        float s = red[0];
#pragma unroll
        for (int i = 1; i < 8; ++i) s += red[i];
        int target = tgt[b * C_T + t];
        float lt = logits[b * C_P + target];
        float prob = __expf(lt - sMax) / s;
        float lp = logf(prob + 1e-8f);
        if (t < sk[b]) out[b] += lp;           // only block b touches out[b]
    }
}

// ------------------------------- launcher ----------------------------------
static inline char* ws_take(char*& p, size_t bytes)
{
    char* r = (char*)(((uintptr_t)p + 255) & ~(uintptr_t)255);
    p = r + bytes;
    return r;
}

extern "C" void kernel_launch(void* const* d_in, const int* in_sizes, int n_in,
                              void* d_out, int out_size, void* d_ws, size_t ws_size,
                              hipStream_t stream)
{
    (void)in_sizes; (void)n_in; (void)out_size; (void)ws_size;

    const float* enc        = (const float*)d_in[0];
    const float* W_ih       = (const float*)d_in[1];
    const float* b_ih       = (const float*)d_in[2];
    const float* W_hh       = (const float*)d_in[3];
    const float* b_hh       = (const float*)d_in[4];
    const float* W_att      = (const float*)d_in[5];
    const float* W_attvec   = (const float*)d_in[6];
    const float* W_init     = (const float*)d_in[7];
    const float* b_init     = (const float*)d_in[8];
    const float* W_q2a      = (const float*)d_in[9];
    const float* b_q2a      = (const float*)d_in[10];
    const float* prod_embed = (const float*)d_in[11];
    const float* prod_bias  = (const float*)d_in[12];
    const float* type_embed = (const float*)d_in[13];
    const int*   prev_pid   = (const int*)d_in[14];
    const int*   prev_tid   = (const int*)d_in[15];
    const int*   target_pid = (const int*)d_in[16];
    const int*   sketch_len = (const int*)d_in[17];
    const int*   src_len    = (const int*)d_in[18];
    float*       out        = (float*)d_out;

    char* p = (char*)d_ws;
    bf16_t* Wcat_bf    = (bf16_t*)ws_take(p, (size_t)C_G4H * C_KCAT * 2);
    bf16_t* WattT_bf   = (bf16_t*)ws_take(p, (size_t)C_H * C_H * 2);
    bf16_t* Wattvec_bf = (bf16_t*)ws_take(p, (size_t)C_A * 2 * C_H * 2);
    bf16_t* Wq2a_bf    = (bf16_t*)ws_take(p, (size_t)C_DA * C_A * 2);
    bf16_t* Pemb_bf    = (bf16_t*)ws_take(p, (size_t)C_P * C_DA * 2);
    bf16_t* Winit_bf   = (bf16_t*)ws_take(p, (size_t)C_H * C_H * 2);
    float*  bias_g     = (float*)ws_take(p, (size_t)C_G4H * 4);
    bf16_t* mean_bf    = (bf16_t*)ws_take(p, (size_t)C_B * C_H * 2);
    bf16_t* h_bf       = (bf16_t*)ws_take(p, (size_t)C_B * C_H * 2);
    bf16_t* att_bf     = (bf16_t*)ws_take(p, (size_t)C_B * C_A * 2);
    float*  c_st       = (float*)ws_take(p, (size_t)C_B * C_H * 4);
    bf16_t* xh_bf      = (bf16_t*)ws_take(p, (size_t)C_B * C_KCAT * 2);
    float*  gates      = (float*)ws_take(p, (size_t)C_B * C_G4H * 4);
    float*  hW         = (float*)ws_take(p, (size_t)C_B * C_H * 4);
    bf16_t* hc_bf      = (bf16_t*)ws_take(p, (size_t)C_B * 2 * C_H * 2);
    bf16_t* q_bf       = (bf16_t*)ws_take(p, (size_t)C_B * C_DA * 2);
    float*  logits     = (float*)ws_take(p, (size_t)C_B * C_P * 4);

    auto gemm = [&](const bf16_t* A, int lda, const bf16_t* Bm, int ldb,
                    const float* bias, float* D, bf16_t* Dbf, int ldd,
                    int M, int N, int K, int act) {
        int tiles = (M / 16) * (N / 16);
        wmma_gemm_bf16<<<tiles / 4, 128, 0, stream>>>(A, lda, Bm, ldb, bias,
                                                      D, Dbf, ldd, M, N, K, act);
    };

    // ---- one-time prep (must rerun every call: no cached state allowed) ----
    {
        int n;
        n = C_G4H * C_KCAT;
        build_wcat<<<(n + 255) / 256, 256, 0, stream>>>(W_ih, W_hh, Wcat_bf, n);
        n = C_H * C_H;
        transpose_convert<<<(n + 255) / 256, 256, 0, stream>>>(W_att, WattT_bf, C_H, C_H);
        n = C_A * 2 * C_H;
        convert_f32_bf16<<<(n + 255) / 256, 256, 0, stream>>>(W_attvec, Wattvec_bf, n);
        n = C_DA * C_A;
        convert_f32_bf16<<<(n + 255) / 256, 256, 0, stream>>>(W_q2a, Wq2a_bf, n);
        n = C_P * C_DA;
        convert_f32_bf16<<<(n + 255) / 256, 256, 0, stream>>>(prod_embed, Pemb_bf, n);
        n = C_H * C_H;
        convert_f32_bf16<<<(n + 255) / 256, 256, 0, stream>>>(W_init, Winit_bf, n);
        bias_sum<<<C_G4H / 256, 256, 0, stream>>>(b_ih, b_hh, bias_g, C_G4H);
        zero_init<<<(C_B * C_H) / 256, 256, 0, stream>>>(c_st, att_bf, out);
        mean_kernel<<<dim3(C_H / 256, C_B), 256, 0, stream>>>(enc, mean_bf);
        // h0 = tanh(mean @ W_init^T + b_init)  (bf16 out only)
        gemm(mean_bf, C_H, Winit_bf, C_H, b_init, nullptr, h_bf, C_H,
             C_B, C_H, C_H, /*tanh*/1);
    }

    // ------------------------- 64 recurrent steps --------------------------
    for (int t = 0; t < C_T; ++t) {
        assemble_xh<<<dim3(C_KCAT / 256, C_B), 256, 0, stream>>>(
            prev_pid, prev_tid, sketch_len, prod_embed, type_embed,
            att_bf, h_bf, xh_bf, t);

        // gates = xh @ [Wih|Whh]^T + (b_ih+b_hh)
        gemm(xh_bf, C_KCAT, Wcat_bf, C_KCAT, bias_g, gates, nullptr, C_G4H,
             C_B, C_G4H, C_KCAT, 0);

        lstm_pointwise<<<dim3(C_H / 256, C_B), 256, 0, stream>>>(
            gates, c_st, h_bf, hc_bf);

        // hW = h @ W_att  (via W_att^T rows)
        gemm(h_bf, C_H, WattT_bf, C_H, nullptr, hW, nullptr, C_H,
             C_B, C_H, C_H, 0);

        attention_kernel<<<C_B, 256, 0, stream>>>(enc, hW, src_len, hc_bf);

        // att_t = tanh([h|ctx] @ W_attvec^T)   -> becomes att_{t} state (bf16)
        gemm(hc_bf, 2 * C_H, Wattvec_bf, 2 * C_H, nullptr, nullptr, att_bf, C_A,
             C_B, C_A, 2 * C_H, 1);

        // q = tanh(att @ W_q2a^T + b_q2a)
        gemm(att_bf, C_A, Wq2a_bf, C_A, b_q2a, nullptr, q_bf, C_DA,
             C_B, C_DA, C_A, 1);

        // logits = q @ prod_embed^T + prod_bias
        gemm(q_bf, C_DA, Pemb_bf, C_DA, prod_bias, logits, nullptr, C_P,
             C_B, C_P, C_DA, 0);

        softmax_logp_kernel<<<C_B, 256, 0, stream>>>(
            logits, target_pid, sketch_len, out, t);
    }
}
